// RGBDSegmentationModel_70592082477492
// MI455X (gfx1250) — compile-verified, matrix operand order checked by reading, not run on hardware
//
#include <hip/hip_runtime.h>
#include <hip/hip_bf16.h>
#include <math.h>

// MI455X / gfx1250 implementation.
// All GEMM-shaped work (attention GEMMs + 3x3 convs via implicit im2col)
// runs through v_wmma_f32_16x16x32_bf16 (bf16 A/B, fp32 accumulate).
// - Attention matrices pad the pixel dim 3600 -> 3648 (57*64): GEMM staging
//   is branch-free (b128 global loads + b128/b16 LDS stores).
// - Conv inputs live in halo-padded images [C][62][72] (zeroed borders) and
//   conv columns use a 64-wide grid (N'=3840), so the im2col gather is ALSO
//   branch-free: one unaligned 16B load per 8 taps, no edge fallbacks, and
//   the k->(ci,dy,dx) split is carried incrementally (no divisions).
//
// Shapes: B=2, C=256, H=W=60, NC=2, OUT=473x473.
// Peak workspace use ~183 MB (S fp32 region is recycled for conv buffers).

typedef __bf16 v16bf __attribute__((ext_vector_type(16)));
typedef __bf16 v8bf  __attribute__((ext_vector_type(8)));
typedef float  v8f   __attribute__((ext_vector_type(8)));

#define HN 60
#define NPIX 3600
#define NP 3648      // padded pixel dim for attention: 57*64
#define NPC 3840     // conv column dim: 64-wide rows * 60
#define PW 72        // halo-padded image width
#define PH 62        // halo-padded image height
#define PST 4464     // PW*PH, channel stride in padded image
#define CCH 256
#define BATCH 2

__device__ inline v16bf ld_frag(const __bf16* p) {
  // two 16-byte LDS loads -> 16 bf16 fragment (per-lane contiguous K)
  v8bf lo = *reinterpret_cast<const v8bf*>(p);
  v8bf hi = *reinterpret_cast<const v8bf*>(p + 8);
  v16bf r;
#pragma unroll
  for (int i = 0; i < 8; ++i) { r[i] = lo[i]; r[i + 8] = hi[i]; }
  return r;
}

__device__ inline v8bf ld8_unaligned(const __bf16* p) {
  v8bf v;
  __builtin_memcpy(&v, p, 16);  // 2-byte aligned 16B load
  return v;
}

// ---------------------------------------------------------------------------
// Branch-free bf16 WMMA GEMM: D(MxN) = A(MxK) * B(KxN), fp32 accumulate.
// REQUIRES: M%64==0, N%64==0, K%32==0, operands padded with zeros.
// a(m,k) = TA ? A[k*lda+m] : A[m*lda+k]
// b(k,n) = TB ? B[n*ldb+k] : B[k*ldb+n]
// Block = 256 threads (8 waves), 64x64 tile, BK=32. blockIdx: x=Ntile,
// y=Mtile, z=batch (per-batch element strides sA,sB,sC).
// ---------------------------------------------------------------------------
template <bool TA, bool TB, bool OUTBF>
__global__ __launch_bounds__(256)
void gemm_wmma(const __bf16* __restrict__ Aall, const __bf16* __restrict__ Ball,
               void* __restrict__ Call, int K,
               int lda, int ldb, int ldc,
               long long sA, long long sB, long long sC) {
  const int bz = blockIdx.z;
  const __bf16* A = Aall + (long long)bz * sA;
  const __bf16* B = Ball + (long long)bz * sB;
  const int Mbase = blockIdx.y * 64;
  const int Nbase = blockIdx.x * 64;

  __shared__ __align__(16) __bf16 As[64][40];  // +8 pad vs bank conflicts
  __shared__ __align__(16) __bf16 Bs[64][40];  // K-major per row/column

  const int tid = threadIdx.x;
  const int wv = tid >> 5, lane = tid & 31;
  const int mrow = (wv >> 2) * 32;   // wave's 32-row slab
  const int ncol = (wv & 3) * 16;    // wave's 16-col slab
  const int fr = lane & 15;          // fragment row/col within tile
  const int fo = (lane >> 4) * 16;   // fragment K offset (lane halves)

  // direct (contiguous-K) staging: 1 thread = 8 consecutive k of one row/col
  const int li  = tid >> 2;          // 0..63
  const int lk0 = (tid & 3) * 8;     // 0,8,16,24
  // transposed staging: 1 thread = 8 consecutive m/n of one k
  const int tk  = tid & 31;          // k lane
  const int ti0 = (tid >> 5) * 8;    // 8-run along m/n
  const int gmA = Mbase + li;        // !TA row
  const int gm0 = Mbase + ti0;       // TA 8-run base
  const int gnB = Nbase + li;        // TB col
  const int gn0 = Nbase + ti0;       // !TB 8-run base

  v8f acc0 = {};
  v8f acc1 = {};

  for (int kb = 0; kb < K; kb += 32) {
    // ---- stage A tile: As[m][k] ----
    if (TA) {
      v8bf v = *reinterpret_cast<const v8bf*>(A + (long long)(kb + tk) * lda + gm0);
#pragma unroll
      for (int u = 0; u < 8; ++u) As[ti0 + u][tk] = v[u];  // reg transpose
    } else {
      v8bf v = *reinterpret_cast<const v8bf*>(A + (long long)gmA * lda + kb + lk0);
      *reinterpret_cast<v8bf*>(&As[li][lk0]) = v;
    }
    // ---- stage B tile: Bs[n][k] ----
    if (TB) {
      v8bf v = *reinterpret_cast<const v8bf*>(B + (long long)gnB * ldb + kb + lk0);
      *reinterpret_cast<v8bf*>(&Bs[li][lk0]) = v;
    } else {
      v8bf v = *reinterpret_cast<const v8bf*>(B + (long long)(kb + tk) * ldb + gn0);
#pragma unroll
      for (int u = 0; u < 8; ++u) Bs[ti0 + u][tk] = v[u];  // reg transpose
    }
    if (kb + 32 < K)  // speculative prefetch of next K tile (global_prefetch_b8)
      __builtin_prefetch(A + (TA ? (long long)(kb + 32) * lda + gm0
                                 : (long long)gmA * lda + kb + 32),
                         0, 1);
    __syncthreads();

    v16bf a0 = ld_frag(&As[mrow + fr][fo]);
    v16bf a1 = ld_frag(&As[mrow + 16 + fr][fo]);
    v16bf bb = ld_frag(&Bs[ncol + fr][fo]);
    acc0 = __builtin_amdgcn_wmma_f32_16x16x32_bf16(false, a0, false, bb,
                                                   (short)0, acc0, false, false);
    acc1 = __builtin_amdgcn_wmma_f32_16x16x32_bf16(false, a1, false, bb,
                                                   (short)0, acc1, false, false);
    __syncthreads();
  }

  const int gn = Nbase + ncol + fr;
  const long long cb = (long long)bz * sC;
#pragma unroll
  for (int r = 0; r < 8; ++r) {
    const int lm  = r + (lane >> 4) * 8;  // C/D layout: lanes16-31 -> M=r+8
    const int gm0o = Mbase + mrow + lm;
    const int gm1o = gm0o + 16;
    if (OUTBF) {
      ((__bf16*)Call)[cb + (long long)gm0o * ldc + gn] = (__bf16)acc0[r];
      ((__bf16*)Call)[cb + (long long)gm1o * ldc + gn] = (__bf16)acc1[r];
    } else {
      ((float*)Call)[cb + (long long)gm0o * ldc + gn] = acc0[r];
      ((float*)Call)[cb + (long long)gm1o * ldc + gn] = acc1[r];
    }
  }
}

// ---------------------------------------------------------------------------
// 3x3 same-pad conv as implicit-im2col WMMA GEMM. Fully branch-free.
// Input: halo-padded images XP[C][PH][PW] (zeroed borders/pads).
// Output columns j in a 64-wide grid: j = y*64+x, N'=NPC; x in [60,64) are
// discarded pad columns. K = Cin*9; (ci,dy,dx) carried incrementally.
// ---------------------------------------------------------------------------
__global__ __launch_bounds__(256)
void conv3x3_wmma(const __bf16* __restrict__ Wt, const __bf16* __restrict__ XPall,
                  float* __restrict__ Yall, int Cin,
                  long long sX, long long sY) {
  const int Kd = Cin * 9;
  const int bz = blockIdx.z;
  const __bf16* XP = XPall + (long long)bz * sX;
  float* Y = Yall + (long long)bz * sY;
  const int Mbase = blockIdx.y * 64;
  const int Nbase = blockIdx.x * 64;

  __shared__ __align__(16) __bf16 As[64][40];
  __shared__ __align__(16) __bf16 Bs[64][40];

  const int tid = threadIdx.x;
  const int wv = tid >> 5, lane = tid & 31;
  const int mrow = (wv >> 2) * 32, ncol = (wv & 3) * 16;
  const int fr = lane & 15, fo = (lane >> 4) * 16;

  // A staging (weights, contiguous K)
  const int li  = tid >> 2;
  const int lk0 = (tid & 3) * 8;
  const int gmA = Mbase + li;
  // B staging (im2col, 8 columns per thread at fixed k)
  const int tk  = tid & 31;
  const int tj0 = (tid >> 5) * 8;
  const int j0  = Nbase + tj0;
  const int y   = j0 >> 6;          // image row (0..59)
  const int x0  = j0 & 63;          // 8-run start within 64-wide row
  const __bf16* XProw = XP + y * PW + x0;  // padded tap base for (ci=0,dy=0,dx=0)

  int ci = tk / 9;                  // compile-time-strength-reduced once
  int rr = tk - 9 * ci;             // rr = dy*3+dx

  v8f acc0 = {};
  v8f acc1 = {};

  for (int kb = 0; kb < Kd; kb += 32) {
    // ---- weights (branch-free) ----
    {
      v8bf v = *reinterpret_cast<const v8bf*>(Wt + (long long)gmA * Kd + kb + lk0);
      *reinterpret_cast<v8bf*>(&As[li][lk0]) = v;
    }
    // ---- im2col (branch-free: halo-padded source) ----
    {
      const int dy = (rr >= 6) ? 2 : ((rr >= 3) ? 1 : 0);
      const int dx = rr - 3 * dy;
      v8bf v = ld8_unaligned(XProw + (long long)ci * PST + dy * PW + dx);
#pragma unroll
      for (int u = 0; u < 8; ++u) Bs[tj0 + u][tk] = v[u];
      // advance (ci, rr) by k += 32  (32 = 3*9 + 5)
      const int t = rr + 5;
      const int c4 = (t >= 9) ? 1 : 0;
      ci += 3 + c4;
      rr = t - (c4 ? 9 : 0);
    }
    if (kb + 32 < Kd)
      __builtin_prefetch(Wt + (long long)gmA * Kd + kb + 32, 0, 1);
    __syncthreads();

    v16bf a0 = ld_frag(&As[mrow + fr][fo]);
    v16bf a1 = ld_frag(&As[mrow + 16 + fr][fo]);
    v16bf bb = ld_frag(&Bs[ncol + fr][fo]);
    acc0 = __builtin_amdgcn_wmma_f32_16x16x32_bf16(false, a0, false, bb,
                                                   (short)0, acc0, false, false);
    acc1 = __builtin_amdgcn_wmma_f32_16x16x32_bf16(false, a1, false, bb,
                                                   (short)0, acc1, false, false);
    __syncthreads();
  }

  const int gn = Nbase + ncol + fr;
#pragma unroll
  for (int r = 0; r < 8; ++r) {
    const int lm  = r + (lane >> 4) * 8;
    const int gm0 = Mbase + mrow + lm;
    Y[(long long)gm0 * NPC + gn] = acc0[r];
    Y[(long long)(gm0 + 16) * NPC + gn] = acc1[r];
  }
}

// ---------------------------------------------------------------------------
// Softmax along last axis of S[b][n][m]: rows stride NP, valid length 3600;
// pad columns written as zero so padded-K GEMM contributions vanish.
// ---------------------------------------------------------------------------
__global__ __launch_bounds__(256)
void softmax_row(const float* __restrict__ S, __bf16* __restrict__ O) {
  const long long r = blockIdx.x;
  const float* row = S + r * (long long)NP;
  __bf16* orow = O + r * (long long)NP;
  __shared__ float red[256];
  const int tid = threadIdx.x;

  float mx = -1e30f;
  for (int j = tid; j < NPIX; j += 256) mx = fmaxf(mx, row[j]);
  red[tid] = mx;
  __syncthreads();
  for (int off = 128; off > 0; off >>= 1) {
    if (tid < off) red[tid] = fmaxf(red[tid], red[tid + off]);
    __syncthreads();
  }
  const float m = red[0];
  __syncthreads();

  float s = 0.f;
  for (int j = tid; j < NPIX; j += 256) s += __expf(row[j] - m);
  red[tid] = s;
  __syncthreads();
  for (int off = 128; off > 0; off >>= 1) {
    if (tid < off) red[tid] += red[tid + off];
    __syncthreads();
  }
  const float inv = 1.f / red[0];

  for (int j = tid; j < NP; j += 256)
    orow[j] = (j < NPIX) ? (__bf16)(__expf(row[j] - m) * inv) : (__bf16)0.f;
}

// Softmax along first axis (over n) of S[b][n][m]; coalesced column walk.
// Pad rows [3600,NP) written as zero.
__global__ __launch_bounds__(256)
void softmax_col(const float* __restrict__ Sall, __bf16* __restrict__ Oall) {
  const int bz = blockIdx.y;
  const float* S = Sall + (long long)bz * NP * NP;
  __bf16* O = Oall + (long long)bz * NP * NP;
  const int j = blockIdx.x * 256 + threadIdx.x;
  if (j >= NP) return;
  float mx = -1e30f;
  for (int i = 0; i < NPIX; ++i) mx = fmaxf(mx, S[(long long)i * NP + j]);
  float s = 0.f;
  for (int i = 0; i < NPIX; ++i) s += __expf(S[(long long)i * NP + j] - mx);
  const float inv = 1.f / s;
  for (int i = 0; i < NPIX; ++i)
    O[(long long)i * NP + j] = (__bf16)(__expf(S[(long long)i * NP + j] - mx) * inv);
  for (int i = NPIX; i < NP; ++i) O[(long long)i * NP + j] = (__bf16)0.f;
}

// ---------------------------------------------------------------------------
// Per-pixel sigmoid gate (1x1 conv over C) using wave32 shfl reduction; writes
// gated attention into channels [0,C) of the halo-padded bf16 image buffer.
// Z has stride NP. One wave per pixel, 8 pixels per block.
// ---------------------------------------------------------------------------
__global__ __launch_bounds__(256)
void gate_store(const float* __restrict__ Zall, const float* __restrict__ gw,
                __bf16* __restrict__ dstall, long long sZ, long long sD) {
  const int bz = blockIdx.y;
  const float* Z = Zall + (long long)bz * sZ;
  __bf16* dst = dstall + (long long)bz * sD;
  const int wv = threadIdx.x >> 5, lane = threadIdx.x & 31;
  const int p = blockIdx.x * 8 + wv;
  const int y = p / HN, x = p % HN;
  const int po = (y + 1) * PW + (x + 1);  // halo offset

  float vr[8];
  float part = 0.f;
#pragma unroll
  for (int j = 0; j < 8; ++j) {
    const int c = lane + 32 * j;
    const float v = Z[(long long)c * NP + p];
    vr[j] = v;
    part += v * gw[c];
  }
#pragma unroll
  for (int m = 16; m > 0; m >>= 1) part += __shfl_xor(part, m, 32);
  const float g = 1.f / (1.f + __expf(-part));
#pragma unroll
  for (int j = 0; j < 8; ++j) {
    const int c = lane + 32 * j;
    dst[(long long)c * PST + po] = (__bf16)(vr[j] * g);
  }
}

// fp32 [B][Cc][3600] -> halo-padded bf16 image [B][Ctot][PH][PW] at Coff.
__global__ __launch_bounds__(256)
void cvt_img(const float* __restrict__ src, __bf16* __restrict__ dst,
             long long n, int Cc, int Ctot, int Coff) {
  const long long idx = (long long)blockIdx.x * 256 + threadIdx.x;
  if (idx >= n) return;
  const int p = (int)(idx % NPIX);
  const long long t = idx / NPIX;
  const int c = (int)(t % Cc);
  const int b = (int)(t / Cc);
  const int y = p / HN, x = p % HN;
  dst[((long long)b * Ctot + Coff + c) * PST + (y + 1) * PW + (x + 1)] =
      (__bf16)src[idx];
}

// fp32 -> bf16 with channel-offset scatter; separate src width / dst stride.
__global__ __launch_bounds__(256)
void cvt_cat(const float* __restrict__ src, __bf16* __restrict__ dst,
             long long n, int Cc, int Np, int Ctot, int Coff, int NpD) {
  const long long idx = (long long)blockIdx.x * 256 + threadIdx.x;
  if (idx >= n) return;
  const int p = (int)(idx % Np);
  const long long t = idx / Np;
  const int c = (int)(t % Cc);
  const int b = (int)(t / Cc);
  dst[((long long)b * Ctot + Coff + c) * NpD + p] = (__bf16)src[idx];
}

// Zero a dword region (for bf16 pad columns / halos).
__global__ __launch_bounds__(256)
void zero_u32(unsigned int* __restrict__ p, long long n) {
  const long long idx = (long long)blockIdx.x * 256 + threadIdx.x;
  if (idx < n) p[idx] = 0u;
}

// Training-mode BatchNorm over (B,H,W) + ReLU, in place (Y stride NPC,
// columns j = y*64+x).
__global__ __launch_bounds__(256)
void bn_relu(float* __restrict__ Y, const float* __restrict__ g,
             const float* __restrict__ bta) {
  const int ch = blockIdx.x;
  const int tid = threadIdx.x;
  __shared__ float s1[256], s2[256];
  const int total = BATCH * NPIX;
  float a = 0.f, q = 0.f;
  for (int idx = tid; idx < total; idx += 256) {
    const int b = idx / NPIX, p = idx % NPIX;
    const int j = (p / HN) * 64 + (p % HN);
    const float v = Y[((long long)b * CCH + ch) * NPC + j];
    a += v;
    q += v * v;
  }
  s1[tid] = a;
  s2[tid] = q;
  __syncthreads();
  for (int off = 128; off > 0; off >>= 1) {
    if (tid < off) { s1[tid] += s1[tid + off]; s2[tid] += s2[tid + off]; }
    __syncthreads();
  }
  const float mean = s1[0] / total;
  const float var = s2[0] / total - mean * mean;
  const float sc = g[ch] * rsqrtf(var + 1e-5f);
  const float sh = bta[ch] - mean * sc;
  for (int idx = tid; idx < total; idx += 256) {
    const int b = idx / NPIX, p = idx % NPIX;
    const int j = (p / HN) * 64 + (p % HN);
    const long long o = ((long long)b * CCH + ch) * NPC + j;
    Y[o] = fmaxf(Y[o] * sc + sh, 0.f);
  }
}

// 1x1 classifier conv (C -> 2) + bias. Y stride NPC (j=y*64+x), out stride NPIX.
__global__ __launch_bounds__(256)
void cls1x1(const float* __restrict__ Y, const float* __restrict__ w,
            const float* __restrict__ bias, float* __restrict__ out) {
  const int idx = blockIdx.x * 256 + threadIdx.x;
  if (idx >= BATCH * NPIX) return;
  const int b = idx / NPIX, p = idx % NPIX;
  const int j = (p / HN) * 64 + (p % HN);
  float a0 = bias[0], a1 = bias[1];
  for (int c = 0; c < CCH; ++c) {
    const float v = Y[((long long)b * CCH + c) * NPC + j];
    a0 += v * w[c];
    a1 += v * w[CCH + c];
  }
  out[((long long)b * 2 + 0) * NPIX + p] = a0;
  out[((long long)b * 2 + 1) * NPIX + p] = a1;
}

// Bilinear 60x60 -> 473x473 (half-pixel centers, edge clamp) + sigmoid.
__global__ __launch_bounds__(256)
void resize_sig(const float* __restrict__ src, float* __restrict__ dst, int planes) {
  const long long idx = (long long)blockIdx.x * 256 + threadIdx.x;
  const long long total = (long long)planes * 473 * 473;
  if (idx >= total) return;
  const int ox = (int)(idx % 473);
  const long long t = idx / 473;
  const int oy = (int)(t % 473);
  const int pl = (int)(t / 473);
  const float* sp = src + (long long)pl * NPIX;
  const float SC = 60.f / 473.f;
  const float fy = (oy + 0.5f) * SC - 0.5f;
  const float fx = (ox + 0.5f) * SC - 0.5f;
  const int iy0 = (int)floorf(fy);
  const int ix0 = (int)floorf(fx);
  const float wy = fy - iy0, wx = fx - ix0;
  const int y0 = min(HN - 1, max(0, iy0)), y1 = min(HN - 1, max(0, iy0 + 1));
  const int x0 = min(HN - 1, max(0, ix0)), x1 = min(HN - 1, max(0, ix0 + 1));
  const float v =
      (1.f - wy) * ((1.f - wx) * sp[y0 * HN + x0] + wx * sp[y0 * HN + x1]) +
      wy * ((1.f - wx) * sp[y1 * HN + x0] + wx * sp[y1 * HN + x1]);
  dst[idx] = 1.f / (1.f + __expf(-v));
}

// ---------------------------------------------------------------------------

extern "C" void kernel_launch(void* const* d_in, const int* in_sizes, int n_in,
                              void* d_out, int out_size, void* d_ws, size_t ws_size,
                              hipStream_t stream) {
  (void)in_sizes; (void)n_in; (void)out_size; (void)ws_size;

  const float* V_a    = (const float*)d_in[0];
  const float* V_b    = (const float*)d_in[1];
  const float* D_a    = (const float*)d_in[2];
  const float* W_e    = (const float*)d_in[3];
  const float* gate_w = (const float*)d_in[4];
  const float* c1w    = (const float*)d_in[5];
  const float* c2w    = (const float*)d_in[6];
  const float* bn1g   = (const float*)d_in[7];
  const float* bn1b   = (const float*)d_in[8];
  const float* bn2g   = (const float*)d_in[9];
  const float* bn2b   = (const float*)d_in[10];
  const float* cl1w   = (const float*)d_in[11];
  const float* cl1b   = (const float*)d_in[12];
  const float* cl2w   = (const float*)d_in[13];
  const float* cl2b   = (const float*)d_in[14];
  float* out = (float*)d_out;

  char* ws = (char*)d_ws;
  // Long-lived regions
  const size_t OFF_S    = 0;                    // S fp32: 2*NP*NP*4 = 106,463,232
  const size_t OFF_SM   = 106463232;            // softmax bf16: 53,231,616
  const size_t OFF_ZA   = OFF_SM + 53231616;    // 2*256*NP*4 = 7,471,104
  const size_t OFF_ZB   = OFF_ZA + 7471104;
  const size_t OFF_VAB  = OFF_ZB + 7471104;     // 2*256*NP*2 = 3,735,552
  const size_t OFF_VBB  = OFF_VAB + 3735552;
  const size_t OFF_WGT  = OFF_VBB + 3735552;    // weighted bf16 [NP][256]
  const size_t OFF_WEB  = OFF_WGT + 3735552;    // 131,072
  const size_t OFF_W1B  = OFF_WEB + 131072;     // 3,538,944
  const size_t OFF_W2B  = OFF_W1B + 3538944;    // 2,359,296 -> end ~183 MB
  // Regions recycled inside the (dead after softmaxes) S area
  const size_t OFF_X1P  = 0;                    // 2*768*PST*2 = 13,713,408
  const size_t OFF_X2P  = OFF_X1P + 13713408;   // 2*512*PST*2 =  9,142,272
  const size_t OFF_Y1   = OFF_X2P + 9142272;    // 2*256*NPC*4 =  7,864,320
  const size_t OFF_Y2   = OFF_Y1 + 7864320;
  const size_t OFF_C1   = OFF_Y2 + 7864320;     // 57,600
  const size_t OFF_C2   = OFF_C1 + 57600;

  float*  S_f   = (float*)(ws + OFF_S);
  __bf16* SM_b  = (__bf16*)(ws + OFF_SM);
  float*  Za    = (float*)(ws + OFF_ZA);
  float*  Zb    = (float*)(ws + OFF_ZB);
  __bf16* Va_b  = (__bf16*)(ws + OFF_VAB);
  __bf16* Vb_b  = (__bf16*)(ws + OFF_VBB);
  __bf16* Wgt_b = (__bf16*)(ws + OFF_WGT);
  __bf16* We_b  = (__bf16*)(ws + OFF_WEB);
  __bf16* W1_b  = (__bf16*)(ws + OFF_W1B);
  __bf16* W2_b  = (__bf16*)(ws + OFF_W2B);
  __bf16* X1P   = (__bf16*)(ws + OFF_X1P);
  __bf16* X2P   = (__bf16*)(ws + OFF_X2P);
  float*  Y1    = (float*)(ws + OFF_Y1);
  float*  Y2    = (float*)(ws + OFF_Y2);
  float*  C1o   = (float*)(ws + OFF_C1);
  float*  C2o   = (float*)(ws + OFF_C2);

  const long long nV  = (long long)BATCH * CCH * NPIX;    // 1,843,200
  const long long nW1 = 256LL * 768 * 9;                  // 1,769,472
  const long long nW2 = 256LL * 512 * 9;                  // 1,179,648
  auto blk = [](long long n) { return (unsigned)((n + 255) / 256); };

  // --- zero Va_b+Vb_b (adjacent) so pad columns are 0 ---------------------
  const long long nZeroDw = (2LL * BATCH * CCH * NP * 2) / 4;  // 1,867,776
  zero_u32<<<blk(nZeroDw), 256, 0, stream>>>((unsigned int*)Va_b, nZeroDw);

  // --- fp32 -> bf16 conversions -------------------------------------------
  cvt_cat<<<blk(nV), 256, 0, stream>>>(V_a, Va_b, nV, CCH, NPIX, CCH, 0, NP);
  cvt_cat<<<blk(nV), 256, 0, stream>>>(V_b, Vb_b, nV, CCH, NPIX, CCH, 0, NP);
  cvt_cat<<<blk(65536), 256, 0, stream>>>(W_e, We_b, 65536, 1, 65536, 1, 0, 65536);
  cvt_cat<<<blk(nW1), 256, 0, stream>>>(c1w, W1_b, nW1, 1, (int)nW1, 1, 0, (int)nW1);
  cvt_cat<<<blk(nW2), 256, 0, stream>>>(c2w, W2_b, nW2, 1, (int)nW2, 1, 0, (int)nW2);

  // --- weighted[b,n,d] = sum_c Va[b,c,n] * We[d,c]   (bf16 out) -----------
  gemm_wmma<true, true, true><<<dim3(4, NP / 64, BATCH), 256, 0, stream>>>(
      Va_b, We_b, (void*)Wgt_b, CCH, NP, CCH, CCH,
      (long long)CCH * NP, 0, (long long)NP * CCH);

  // --- S[b,n,m] = weighted[b,n,:] . Vb[b,:,m]   (fp32 out) ----------------
  gemm_wmma<false, false, false><<<dim3(NP / 64, NP / 64, BATCH), 256, 0, stream>>>(
      Wgt_b, Vb_b, (void*)S_f, CCH, CCH, NP, NP,
      (long long)NP * CCH, (long long)CCH * NP, (long long)NP * NP);

  // --- Z_a = Vb * rowSM(S)^T ----------------------------------------------
  softmax_row<<<BATCH * NP, 256, 0, stream>>>(S_f, SM_b);
  gemm_wmma<false, true, false><<<dim3(NP / 64, 4, BATCH), 256, 0, stream>>>(
      Vb_b, SM_b, (void*)Za, NP, NP, NP, NP,
      (long long)CCH * NP, (long long)NP * NP, (long long)CCH * NP);

  // --- Z_b = Va * colSM(S)  (reuse SM region) -----------------------------
  softmax_col<<<dim3((NP + 255) / 256, BATCH), 256, 0, stream>>>(S_f, SM_b);
  gemm_wmma<false, false, false><<<dim3(NP / 64, 4, BATCH), 256, 0, stream>>>(
      Va_b, SM_b, (void*)Zb, NP, NP, NP, NP,
      (long long)CCH * NP, (long long)NP * NP, (long long)CCH * NP);

  // --- zero halo-padded conv inputs (X1P+X2P adjacent, S now dead) --------
  const long long nZeroXP = (13713408LL + 9142272LL) / 4;  // 5,713,920 dwords
  zero_u32<<<blk(nZeroXP), 256, 0, stream>>>((unsigned int*)X1P, nZeroXP);

  // --- sigmoid gates -> padded image buffers (channels [0,256)) -----------
  gate_store<<<dim3(450, BATCH), 256, 0, stream>>>(
      Za, gate_w, X1P, (long long)CCH * NP, 768LL * PST);
  gate_store<<<dim3(450, BATCH), 256, 0, stream>>>(
      Zb, gate_w, X2P, (long long)CCH * NP, 512LL * PST);
  // concat rest: V_a, D_a -> X1P; V_b -> X2P
  cvt_img<<<blk(nV), 256, 0, stream>>>(V_a, X1P, nV, CCH, 768, 256);
  cvt_img<<<blk(nV), 256, 0, stream>>>(D_a, X1P, nV, CCH, 768, 512);
  cvt_img<<<blk(nV), 256, 0, stream>>>(V_b, X2P, nV, CCH, 512, 256);

  // --- 3x3 convs as implicit-GEMM WMMA (branch-free) ----------------------
  conv3x3_wmma<<<dim3(NPC / 64, 4, BATCH), 256, 0, stream>>>(
      W1_b, X1P, Y1, 768, 768LL * PST, (long long)CCH * NPC);
  conv3x3_wmma<<<dim3(NPC / 64, 4, BATCH), 256, 0, stream>>>(
      W2_b, X2P, Y2, 512, 512LL * PST, (long long)CCH * NPC);

  // --- BN (train) + ReLU ---------------------------------------------------
  bn_relu<<<CCH, 256, 0, stream>>>(Y1, bn1g, bn1b);
  bn_relu<<<CCH, 256, 0, stream>>>(Y2, bn2g, bn2b);

  // --- 1x1 classifiers -----------------------------------------------------
  cls1x1<<<blk(BATCH * NPIX), 256, 0, stream>>>(Y1, cl1w, cl1b, C1o);
  cls1x1<<<blk(BATCH * NPIX), 256, 0, stream>>>(Y2, cl2w, cl2b, C2o);

  // --- bilinear upsample + sigmoid -> d_out --------------------------------
  const long long outN = (long long)BATCH * 2 * 473 * 473;  // 894,916
  resize_sig<<<blk(outN), 256, 0, stream>>>(C1o, out, BATCH * 2);
  resize_sig<<<blk(outN), 256, 0, stream>>>(C2o, out + outN, BATCH * 2);
}
